// Attention_54958401519946
// MI455X (gfx1250) — matrix operand order repaired and checked by reading, not compiled, not run on hardware
//
#include <hip/hip_runtime.h>
#include <hip/hip_bf16.h>

typedef _Float16 h16;
typedef __attribute__((ext_vector_type(16))) _Float16 v16h;
typedef __attribute__((ext_vector_type(8)))  _Float16 v8h;
typedef __attribute__((ext_vector_type(8)))  float    v8f;
typedef __attribute__((ext_vector_type(4)))  float    v4f;

#define BATCH   8
#define SEQ     1024
#define DIM     768
#define HEADS   12
#define DHEAD   64
#define NQKV    2304          // 3*HEADS*DHEAD
#define TOK     (BATCH*SEQ)   // 8192

union V16u { v16h v; v8h h[2]; };
union H2u  { h16 h[2]; unsigned u; };

__device__ __forceinline__ v8f zero8() {
  v8f z;
#pragma unroll
  for (int i = 0; i < 8; ++i) z[i] = 0.0f;
  return z;
}

// A fragment (16x32 f16, MxK). base -> (row0, k0) of row-major LDS tile.
// lane r holds row M=r; half hf: elems 0..7 K=hf*8+0..7, elems 8..15 K=16+hf*8+0..7
__device__ __forceinline__ v16h load_a_frag(const h16* base, int stride, int lane) {
  const int r = lane & 15, hf = lane >> 4;
  const h16* p = base + r * stride + hf * 8;
  V16u u;
  u.h[0] = *(const v8h*)p;
  u.h[1] = *(const v8h*)(p + 16);
  return u.v;
}

// B fragment (32x16 f16, KxN) from K-MAJOR staging [n][k]: lane r = col N;
// elems 0..7 K=hf*8+0..7, elems 8..15 K=16+hf*8+0..7
__device__ __forceinline__ v16h load_b_frag(const h16* base, int stride, int lane) {
  const int r = lane & 15, hf = lane >> 4;
  const h16* p = base + r * stride + hf * 8;
  V16u u;
  u.h[0] = *(const v8h*)p;
  u.h[1] = *(const v8h*)(p + 16);
  return u.v;
}

__device__ __forceinline__ v8f wmma_f16(v16h a, v16h b, v8f c) {
  return __builtin_amdgcn_wmma_f32_16x16x32_f16(false, a, false, b, (short)0, c,
                                                false, false);
}

// Async global->LDS 16B copy (ASYNCcnt-tracked). Low 32 bits of a generic
// pointer to __shared__ are the LDS byte address (ISA 10.2 aperture rules).
__device__ __forceinline__ void async_copy16(const h16* gptr, h16* lptr) {
  unsigned lds = (unsigned)(uintptr_t)lptr;
  asm volatile("global_load_async_to_lds_b128 %0, %1, off"
               :: "v"(lds), "v"(gptr) : "memory");
}
__device__ __forceinline__ void wait_async0() {
  asm volatile("s_wait_asynccnt 0" ::: "memory");
}

// ---------------------------------------------------------------------------
// Kernel 1: QKV = x @ W_qkv, rotary on Q/K, scale folded into Q.
// Double-buffered LDS, register-staged next tile. Grid (64,18), 256 thr.
// ---------------------------------------------------------------------------
__global__ void __launch_bounds__(256)
qkv_rope_kernel(const float* __restrict__ x, const float* __restrict__ pe,
                const float* __restrict__ W, h16* __restrict__ qws,
                h16* __restrict__ kws, h16* __restrict__ vws) {
  __shared__ __align__(16) h16 sA[2][128 * 40];   // row-major [m][k]
  __shared__ __align__(16) h16 sB[2][128 * 40];   // k-major   [n][k]

  const int tid  = threadIdx.x;
  const int lane = tid & 31;
  const int wave = tid >> 5;
  const int wm   = wave & 3;
  const int wn   = wave >> 2;
  const int bm   = blockIdx.x * 128;
  const int bn   = blockIdx.y * 128;

  const int arow = tid & 127;
  const int acb  = (tid >> 7) * 16;          // 0 or 16
  const int bk2  = (tid & 15) * 2;           // even k in 0..30
  const int bnb  = (tid >> 4) * 8;           // n base 0..120

  v8f acc[2][4];
#pragma unroll
  for (int i = 0; i < 2; ++i)
#pragma unroll
    for (int j = 0; j < 4; ++j) acc[i][j] = zero8();

  v4f   af[4];
  float bf0[8], bf1[8];

  auto g_load = [&](int k0) {
    const v4f* g4 = (const v4f*)(x + (size_t)(bm + arow) * DIM + k0 + acb);
    af[0] = g4[0]; af[1] = g4[1]; af[2] = g4[2]; af[3] = g4[3];
    const float* g0 = W + (size_t)(k0 + bk2) * NQKV + bn + bnb;
    const float* g1 = g0 + NQKV;
#pragma unroll
    for (int i = 0; i < 8; ++i) { bf0[i] = g0[i]; bf1[i] = g1[i]; }
  };
  auto s_store = [&](int buf) {
    v8h o0, o1;
#pragma unroll
    for (int i = 0; i < 4; ++i) { o0[i] = (h16)af[0][i]; o0[4 + i] = (h16)af[1][i]; }
#pragma unroll
    for (int i = 0; i < 4; ++i) { o1[i] = (h16)af[2][i]; o1[4 + i] = (h16)af[3][i]; }
    *(v8h*)&sA[buf][arow * 40 + acb]     = o0;
    *(v8h*)&sA[buf][arow * 40 + acb + 8] = o1;
#pragma unroll
    for (int i = 0; i < 8; ++i) {
      H2u p; p.h[0] = (h16)bf0[i]; p.h[1] = (h16)bf1[i];
      *(unsigned*)&sB[buf][(bnb + i) * 40 + bk2] = p.u;
    }
  };

  g_load(0);
  s_store(0);
  __syncthreads();

  int cur = 0;
  for (int k0 = 0; k0 < DIM; k0 += 32) {
    const bool more = (k0 + 32) < DIM;
    if (more) {
      g_load(k0 + 32);                       // overlap with WMMA below
      if (k0 + 64 < DIM)
        __builtin_prefetch(x + (size_t)(bm + arow) * DIM + k0 + 64 + acb, 0, 3);
    }
    v16h a0 = load_a_frag(&sA[cur][(wm * 32) * 40], 40, lane);
    v16h a1 = load_a_frag(&sA[cur][(wm * 32 + 16) * 40], 40, lane);
#pragma unroll
    for (int nt = 0; nt < 4; ++nt) {
      v16h bf = load_b_frag(&sB[cur][(wn * 64 + nt * 16) * 40], 40, lane);
      acc[0][nt] = wmma_f16(a0, bf, acc[0][nt]);
      acc[1][nt] = wmma_f16(a1, bf, acc[1][nt]);
    }
    if (more) s_store(cur ^ 1);
    __syncthreads();
    cur ^= 1;
  }

  // epilogue: rotary + scatter f16
  const int r = lane & 15, hf = lane >> 4;
#pragma unroll
  for (int mt = 0; mt < 2; ++mt) {
#pragma unroll
    for (int nt = 0; nt < 4; ++nt) {
      const int M0    = bm + wm * 32 + mt * 16;
      const int j     = bn + wn * 64 + nt * 16 + r;
      const int which = j / DIM;             // 0=q 1=k 2=v (wave-uniform)
      const int hd    = j - which * DIM;
      const int head  = hd >> 6;
      const int d     = hd & 63;
      h16* dst = (which == 0) ? qws : ((which == 1) ? kws : vws);
#pragma unroll
      for (int i = 0; i < 8; ++i) {
        const int m    = M0 + hf * 8 + i;
        const int pos  = m & (SEQ - 1);
        const int bidx = m >> 10;
        float val      = acc[mt][nt][i];
        float partner  = __shfl_xor(val, 1); // element at d^1
        if (which < 2) {
          const float sinv = pe[pos * 64 + (d >> 1)];
          const float cosv = pe[pos * 64 + 32 + (d >> 1)];
          const float rot  = (d & 1) ? partner : -partner;
          val = val * cosv + rot * sinv;
          if (which == 0) val *= 0.125f;     // DIM_HEAD^-0.5
        }
        const size_t idx =
            ((size_t)(bidx * HEADS + head) * SEQ + pos) * DHEAD + d;
        dst[idx] = (h16)val;
      }
    }
  }
}

// ---------------------------------------------------------------------------
// Kernel 2: flash attention per (b, h, 128-query block); Q/K staged via
// async-to-LDS, V transposed via packed dword stores. Grid (8,12,8), 256 thr.
// ---------------------------------------------------------------------------
__global__ void __launch_bounds__(256)
attn_kernel(const h16* __restrict__ qws, const h16* __restrict__ kws,
            const h16* __restrict__ vws, h16* __restrict__ ows) {
  __shared__ __align__(16) h16 sQ[128 * 72];     // [qrow][d]
  __shared__ __align__(16) h16 sK[64 * 72];      // [kpos][d]
  __shared__ __align__(16) h16 sVt[64 * 72];     // [d][kpos]
  __shared__ __align__(16) h16 sP[8 * 16 * 72];  // [qrow][kpos] per-wave strip

  const int tid  = threadIdx.x;
  const int lane = tid & 31;
  const int wave = tid >> 5;
  const int b    = blockIdx.x;
  const int h    = blockIdx.y;
  const int qb   = blockIdx.z;
  const int r    = lane & 15, hf = lane >> 4;

  const size_t headoff = (size_t)(b * HEADS + h) * SEQ * DHEAD;

  // stage Q block (128x64) via async DMA: 4 x 16B chunks per thread
#pragma unroll
  for (int jj = 0; jj < 4; ++jj) {
    const int c    = tid + 256 * jj;             // 0..1023
    const int row  = c >> 3;
    const int colh = (c & 7) * 8;
    async_copy16(qws + headoff + (size_t)(qb * 128 + row) * DHEAD + colh,
                 &sQ[row * 72 + colh]);
  }

  float mrun[8], lrun[8];
#pragma unroll
  for (int i = 0; i < 8; ++i) { mrun[i] = -1e30f; lrun[i] = 0.0f; }
  v8f oacc[4];
#pragma unroll
  for (int i = 0; i < 4; ++i) oacc[i] = zero8();

  const int vrow = (tid & 31) * 2;               // even kpos 0..62
  const int vcb  = (tid >> 5) * 8;               // d base 0..56

  for (int kb = 0; kb < SEQ / 64; ++kb) {
    // K block via async DMA: 2 x 16B chunks per thread
#pragma unroll
    for (int jj = 0; jj < 2; ++jj) {
      const int c    = tid + 256 * jj;           // 0..511
      const int row  = c >> 3;
      const int colh = (c & 7) * 8;
      async_copy16(kws + headoff + (size_t)(kb * 64 + row) * DHEAD + colh,
                   &sK[row * 72 + colh]);
    }
    // V block transposed: pack kpos pairs -> dword stores
    {
      const h16* gv0 = vws + headoff + (size_t)(kb * 64 + vrow) * DHEAD + vcb;
      const h16* gv1 = gv0 + DHEAD;
#pragma unroll
      for (int i = 0; i < 8; ++i) {
        H2u p; p.h[0] = gv0[i]; p.h[1] = gv1[i];
        *(unsigned*)&sVt[(vcb + i) * 72 + vrow] = p.u;
      }
    }
    wait_async0();
    __syncthreads();

    // S strip = Q[wave*16..+16) @ K^T (16x64 fp32)
    v8f s[4];
#pragma unroll
    for (int nt = 0; nt < 4; ++nt) s[nt] = zero8();
#pragma unroll
    for (int ks = 0; ks < 2; ++ks) {
      v16h a = load_a_frag(sQ + wave * 16 * 72 + ks * 32, 72, lane);
#pragma unroll
      for (int nt = 0; nt < 4; ++nt) {
        v16h bf = load_b_frag(sK + nt * 16 * 72 + ks * 32, 72, lane);
        s[nt] = wmma_f16(a, bf, s[nt]);
      }
    }

    // online softmax (per-row stats redundant across each 16-lane half)
#pragma unroll
    for (int i = 0; i < 8; ++i) {
      float rmax = fmaxf(fmaxf(s[0][i], s[1][i]), fmaxf(s[2][i], s[3][i]));
#pragma unroll
      for (int off = 1; off < 16; off <<= 1) rmax = fmaxf(rmax, __shfl_xor(rmax, off));
      const float mnew = fmaxf(mrun[i], rmax);
      const float corr = __expf(mrun[i] - mnew);
      mrun[i] = mnew;
      const int prow = (wave * 16 + hf * 8 + i) * 72;
      float rsum = 0.0f;
#pragma unroll
      for (int nt = 0; nt < 4; ++nt) {
        const float p = __expf(s[nt][i] - mnew);
        rsum += p;
        sP[prow + nt * 16 + r] = (h16)p;
      }
#pragma unroll
      for (int off = 1; off < 16; off <<= 1) rsum += __shfl_xor(rsum, off);
      lrun[i] = lrun[i] * corr + rsum;
#pragma unroll
      for (int ot = 0; ot < 4; ++ot) oacc[ot][i] *= corr;
    }
    __syncthreads();   // sP visible for A-fragment reload

    // O strip += P @ V
#pragma unroll
    for (int ks = 0; ks < 2; ++ks) {
      v16h a = load_a_frag(sP + wave * 16 * 72 + ks * 32, 72, lane);
#pragma unroll
      for (int ot = 0; ot < 4; ++ot) {
        v16h bf = load_b_frag(sVt + ot * 16 * 72 + ks * 32, 72, lane);
        oacc[ot] = wmma_f16(a, bf, oacc[ot]);
      }
    }
    __syncthreads();   // before next K/V overwrite
  }

  // normalize + store O f16 into [b][n][heads*64]
#pragma unroll
  for (int i = 0; i < 8; ++i) {
    const float inv  = 1.0f / lrun[i];
    const int qpos   = qb * 128 + wave * 16 + hf * 8 + i;
    const size_t off = ((size_t)b * SEQ + qpos) * DIM + h * DHEAD;
#pragma unroll
    for (int ot = 0; ot < 4; ++ot)
      ows[off + ot * 16 + r] = (h16)(oacc[ot][i] * inv);
  }
}

// ---------------------------------------------------------------------------
// Kernel 3: out = O @ W_out + b_out (fp32). Double-buffered. Grid (64,6).
// ---------------------------------------------------------------------------
__global__ void __launch_bounds__(256)
out_proj_kernel(const h16* __restrict__ ows, const float* __restrict__ W,
                const float* __restrict__ bias, float* __restrict__ out) {
  __shared__ __align__(16) h16 sA[2][128 * 40];
  __shared__ __align__(16) h16 sB[2][128 * 40];

  const int tid  = threadIdx.x;
  const int lane = tid & 31;
  const int wave = tid >> 5;
  const int wm   = wave & 3;
  const int wn   = wave >> 2;
  const int bm   = blockIdx.x * 128;
  const int bn   = blockIdx.y * 128;

  const int arow = tid & 127;
  const int acb  = (tid >> 7) * 16;
  const int bk2  = (tid & 15) * 2;
  const int bnb  = (tid >> 4) * 8;

  v8f acc[2][4];
#pragma unroll
  for (int i = 0; i < 2; ++i)
#pragma unroll
    for (int j = 0; j < 4; ++j) acc[i][j] = zero8();

  v8h   ah0, ah1;
  float bf0[8], bf1[8];

  auto g_load = [&](int k0) {
    const h16* gp = ows + (size_t)(bm + arow) * DIM + k0 + acb;
    ah0 = *(const v8h*)gp;
    ah1 = *(const v8h*)(gp + 8);
    const float* g0 = W + (size_t)(k0 + bk2) * DIM + bn + bnb;
    const float* g1 = g0 + DIM;
#pragma unroll
    for (int i = 0; i < 8; ++i) { bf0[i] = g0[i]; bf1[i] = g1[i]; }
  };
  auto s_store = [&](int buf) {
    *(v8h*)&sA[buf][arow * 40 + acb]     = ah0;
    *(v8h*)&sA[buf][arow * 40 + acb + 8] = ah1;
#pragma unroll
    for (int i = 0; i < 8; ++i) {
      H2u p; p.h[0] = (h16)bf0[i]; p.h[1] = (h16)bf1[i];
      *(unsigned*)&sB[buf][(bnb + i) * 40 + bk2] = p.u;
    }
  };

  g_load(0);
  s_store(0);
  __syncthreads();

  int cur = 0;
  for (int k0 = 0; k0 < DIM; k0 += 32) {
    const bool more = (k0 + 32) < DIM;
    if (more) g_load(k0 + 32);
    v16h a0 = load_a_frag(&sA[cur][(wm * 32) * 40], 40, lane);
    v16h a1 = load_a_frag(&sA[cur][(wm * 32 + 16) * 40], 40, lane);
#pragma unroll
    for (int nt = 0; nt < 4; ++nt) {
      v16h bf = load_b_frag(&sB[cur][(wn * 64 + nt * 16) * 40], 40, lane);
      acc[0][nt] = wmma_f16(a0, bf, acc[0][nt]);
      acc[1][nt] = wmma_f16(a1, bf, acc[1][nt]);
    }
    if (more) s_store(cur ^ 1);
    __syncthreads();
    cur ^= 1;
  }

  const int r = lane & 15, hf = lane >> 4;
#pragma unroll
  for (int mt = 0; mt < 2; ++mt) {
#pragma unroll
    for (int nt = 0; nt < 4; ++nt) {
      const int j = bn + wn * 64 + nt * 16 + r;
      const float bj = bias[j];
#pragma unroll
      for (int i = 0; i < 8; ++i) {
        const int m = bm + wm * 32 + mt * 16 + hf * 8 + i;
        out[(size_t)m * DIM + j] = acc[mt][nt][i] + bj;
      }
    }
  }
}

// ---------------------------------------------------------------------------
extern "C" void kernel_launch(void* const* d_in, const int* in_sizes, int n_in,
                              void* d_out, int out_size, void* d_ws,
                              size_t ws_size, hipStream_t stream) {
  const float* x    = (const float*)d_in[0];  // (8,1024,768)
  const float* pe   = (const float*)d_in[1];  // (1,1024,64)
  const float* Wqkv = (const float*)d_in[2];  // (768,2304)
  const float* Wout = (const float*)d_in[3];  // (768,768)
  const float* bout = (const float*)d_in[4];  // (768,)
  float* out = (float*)d_out;

  const size_t HSZ = (size_t)BATCH * HEADS * SEQ * DHEAD;
  h16* qws = (h16*)d_ws;
  h16* kws = qws + HSZ;
  h16* vws = kws + HSZ;
  h16* ows = vws + HSZ;                       // (8192, 768) f16

  qkv_rope_kernel<<<dim3(TOK / 128, NQKV / 128), 256, 0, stream>>>(
      x, pe, Wqkv, qws, kws, vws);
  attn_kernel<<<dim3(BATCH, HEADS, SEQ / 128), 256, 0, stream>>>(qws, kws, vws,
                                                                 ows);
  out_proj_kernel<<<dim3(TOK / 128, DIM / 128), 256, 0, stream>>>(ows, Wout,
                                                                  bout, out);
}
